// BitLinear_21964462752075
// MI455X (gfx1250) — matrix operand order repaired and checked by reading, not compiled, not run on hardware
//
#include <hip/hip_runtime.h>

typedef __attribute__((ext_vector_type(16))) _Float16 v16h;
typedef __attribute__((ext_vector_type(8)))  _Float16 f16x8;
typedef __attribute__((ext_vector_type(8)))  float    v8f;

#define DIM   4096
#define NROWS 16384          // 8 * 2048
#define BM    128
#define BN    256
#define BK    32
#define LDSB  80             // bytes per LDS row: 64B data + 16B pad (bank-conflict free)

// ---- CDNA5 async global->LDS copy (ASYNCcnt path), per-lane 16B ----
__device__ __forceinline__ void async_b128(const void* g, const void* lds_generic) {
    unsigned loff = (unsigned)(unsigned long long)lds_generic;  // low 32b of flat addr = LDS offset
    asm volatile("global_load_async_to_lds_b128 %0, %1, off"
                 :: "v"(loff), "v"(g)
                 : "memory");
}
__device__ __forceinline__ void wait_async0() {
    asm volatile("s_wait_asynccnt 0" ::: "memory");
}

// ---------------- init ----------------
__global__ void init_kernel(unsigned* absmax) {
    *absmax = 0u;            // bits of +0.0f; atomicMax over fabs bits is monotone
}

// ------------- sign(W) -> f16, + Frobenius partial sums -------------
__global__ __launch_bounds__(256) void prep_kernel(const float* __restrict__ W,
                                                   _Float16* __restrict__ Bh,
                                                   float* __restrict__ partial) {
    __shared__ float red[256];
    const int t = threadIdx.x;
    const size_t base = (size_t)blockIdx.x * DIM;
    float acc = 0.f;
#pragma unroll
    for (int s = 0; s < 16; ++s) {
        size_t idx = base + t + s * 256;
        float w = W[idx];
        float sg = (w > 0.f) ? 1.f : ((w < 0.f) ? -1.f : 0.f);
        Bh[idx] = (_Float16)sg;
        acc += w * w;
    }
    red[t] = acc;
    __syncthreads();
    for (int off = 128; off > 0; off >>= 1) {
        if (t < off) red[t] += red[t + off];
        __syncthreads();
    }
    if (t == 0) partial[blockIdx.x] = red[0];
}

// ------------- deterministic final reduce -> ||W||_F -------------
__global__ void frob_reduce_kernel(const float* __restrict__ partial, float* __restrict__ frob) {
    __shared__ float red[256];
    const int t = threadIdx.x;
    float acc = 0.f;
#pragma unroll
    for (int s = 0; s < 16; ++s) acc += partial[t + s * 256];
    red[t] = acc;
    __syncthreads();
    for (int off = 128; off > 0; off >>= 1) {
        if (t < off) red[t] += red[t + off];
        __syncthreads();
    }
    if (t == 0) frob[0] = sqrtf(red[0]);
}

// ------------- LayerNorm (one row per block) -> f16 activations -------------
__global__ __launch_bounds__(256) void ln_kernel(const float* __restrict__ x,
                                                 const float* __restrict__ lw,
                                                 const float* __restrict__ lb,
                                                 _Float16* __restrict__ A) {
    __shared__ float r1[256], r2[256];
    const int t = threadIdx.x;
    const size_t base = (size_t)blockIdx.x * DIM;
    float v[16];
    float s = 0.f, s2 = 0.f;
#pragma unroll
    for (int i = 0; i < 16; ++i) {
        float u = x[base + t + i * 256];
        v[i] = u; s += u; s2 += u * u;
    }
    r1[t] = s; r2[t] = s2;
    __syncthreads();
    for (int off = 128; off > 0; off >>= 1) {
        if (t < off) { r1[t] += r1[t + off]; r2[t] += r2[t + off]; }
        __syncthreads();
    }
    const float mu  = r1[0] * (1.f / DIM);
    const float var = r2[0] * (1.f / DIM) - mu * mu;
    const float rs  = rsqrtf(var + 1e-5f);
#pragma unroll
    for (int i = 0; i < 16; ++i) {
        int col = t + i * 256;
        float xn = (v[i] - mu) * rs * lw[col] + lb[col];
        A[base + col] = (_Float16)xn;
    }
}

// ------------- WMMA GEMM: y = A(f16) x sign(W)^T + bias, track |y| max -------------
// Block tile 128(M) x 256(N), BK=32, double-buffered LDS filled by async DMA.
// 8 waves as 2(M) x 4(N); each wave computes 64x64 via 4x4 WMMA 16x16x32 tiles.
// K loop unrolled by 2 so the LDS buffer parity is static (no runtime buffer select).
__global__ __launch_bounds__(256) void gemm_kernel(const _Float16* __restrict__ A,
                                                   const _Float16* __restrict__ Bh,
                                                   const float* __restrict__ bias,
                                                   float* __restrict__ y,
                                                   unsigned* __restrict__ absmax) {
    __shared__ unsigned char ldsA[2][BM * LDSB];
    __shared__ unsigned char ldsB[2][BN * LDSB];
    __shared__ float redmax[256];

    const int t     = threadIdx.x;
    const int lane  = t & 31;
    const int wave  = t >> 5;       // 0..7
    const int waveM = wave >> 2;    // 0..1  -> 64 rows each
    const int waveN = wave & 3;     // 0..3  -> 64 cols each
    const int    nBase = blockIdx.x * BN;
    const size_t mBase = (size_t)blockIdx.y * BM;

    // A staging: 2 threads per tile row, 32B each. B staging: 1 thread per row, 64B.
    const int aRow  = t >> 1;
    const int aHalf = (t & 1) * 16;                 // f16 offset within 64B row
    const _Float16* gA = A  + (mBase + aRow) * DIM + aHalf;
    const _Float16* gB = Bh + (size_t)(nBase + t) * DIM;

    v8f acc[4][4] = {};

#define ISSUE_TILE(buf, kb)                                             \
    do {                                                                \
        const _Float16* ga = gA + (kb);                                 \
        const unsigned char* la = &ldsA[buf][aRow * LDSB + aHalf * 2];  \
        async_b128(ga,     la);                                         \
        async_b128(ga + 8, la + 16);                                    \
        const _Float16* gb = gB + (kb);                                 \
        const unsigned char* lb = &ldsB[buf][t * LDSB];                 \
        async_b128(gb,      lb);                                        \
        async_b128(gb + 8,  lb + 16);                                   \
        async_b128(gb + 16, lb + 32);                                   \
        async_b128(gb + 24, lb + 48);                                   \
    } while (0)

    // A fragments: lanes 0-15 hold K0..7 / K16..23; lanes 16-31 hold K8..15 / K24..31
    // B fragments: lanes 0-15 hold K0..7 / K16..23 of col n; lanes 16-31 hold K8..15 / K24..31
#define COMPUTE_TILE(buf)                                                                        \
    do {                                                                                         \
        v16h af[4];                                                                              \
        _Pragma("unroll")                                                                        \
        for (int i = 0; i < 4; ++i) {                                                            \
            int row = waveM * 64 + i * 16 + (lane & 15);                                         \
            const _Float16* p = (const _Float16*)&ldsA[buf][row * LDSB];                         \
            int off = (lane < 16) ? 0 : 8;                                                       \
            f16x8 lo = *(const f16x8*)(p + off);                                                 \
            f16x8 hi = *(const f16x8*)(p + off + 16);                                            \
            af[i] = __builtin_shufflevector(lo, hi, 0,1,2,3,4,5,6,7,8,9,10,11,12,13,14,15);      \
        }                                                                                        \
        v16h bf[4];                                                                              \
        _Pragma("unroll")                                                                        \
        for (int j = 0; j < 4; ++j) {                                                            \
            int col = waveN * 64 + j * 16 + (lane & 15);                                         \
            const _Float16* p = (const _Float16*)&ldsB[buf][col * LDSB];                         \
            int off = (lane < 16) ? 0 : 8;                                                       \
            f16x8 lo = *(const f16x8*)(p + off);                                                 \
            f16x8 hi = *(const f16x8*)(p + off + 16);                                            \
            bf[j] = __builtin_shufflevector(lo, hi, 0,1,2,3,4,5,6,7,8,9,10,11,12,13,14,15);      \
        }                                                                                        \
        _Pragma("unroll")                                                                        \
        for (int i = 0; i < 4; ++i)                                                              \
            _Pragma("unroll")                                                                    \
            for (int j = 0; j < 4; ++j)                                                          \
                acc[i][j] = __builtin_amdgcn_wmma_f32_16x16x32_f16(                              \
                    false, af[i], false, bf[j], (short)0, acc[i][j], false, false);              \
    } while (0)

    ISSUE_TILE(0, 0);

    for (int kb = 0; kb < DIM; kb += 2 * BK) {
        // ---- half 1: consume buf0, DMA next tile into buf1 ----
        wait_async0();       // this wave's async writes into buf0 have landed in LDS
        __syncthreads();     // all waves landed; all reads of buf1 (prev iter) done
        ISSUE_TILE(1, kb + BK);                 // kb+BK < DIM always (DIM/BK is even)
        COMPUTE_TILE(0);

        // ---- half 2: consume buf1, DMA next tile into buf0 ----
        wait_async0();
        __syncthreads();
        if (kb + 2 * BK < DIM) ISSUE_TILE(0, kb + 2 * BK);
        COMPUTE_TILE(1);
    }
#undef ISSUE_TILE
#undef COMPUTE_TILE

    // epilogue: bias add, store y, block-local absmax
    float lmax = 0.f;
#pragma unroll
    for (int i = 0; i < 4; ++i) {
#pragma unroll
        for (int j = 0; j < 4; ++j) {
            int col = nBase + waveN * 64 + j * 16 + (lane & 15);
            float bc = bias[col];
            size_t row0 = mBase + waveM * 64 + i * 16 + ((lane < 16) ? 0 : 8);
#pragma unroll
            for (int r = 0; r < 8; ++r) {
                float v = acc[i][j][r] + bc;
                y[(row0 + r) * DIM + col] = v;
                lmax = fmaxf(lmax, fabsf(v));
            }
        }
    }
    redmax[t] = lmax;
    __syncthreads();
    for (int off = 128; off > 0; off >>= 1) {
        if (t < off) redmax[t] = fmaxf(redmax[t], redmax[t + off]);
        __syncthreads();
    }
    if (t == 0) atomicMax(absmax, __float_as_uint(redmax[0]));
}

// ------------- quantize/dequant + rescale, in place on d_out -------------
__global__ __launch_bounds__(256) void finalize_kernel(float* __restrict__ y,
                                                       const unsigned* __restrict__ absmax,
                                                       const float* __restrict__ frob) {
    const float amax  = __uint_as_float(*absmax);
    const float scale = 127.f / amax;
    const float post  = (amax * (1.f / 127.f)) * frob[0] * 64.f;  // 1/scale * ||W||_F * sqrt(4096)
    size_t i = ((size_t)blockIdx.x * 256 + threadIdx.x) * 4;
    float4 v = *(float4*)(y + i);
    v.x = rintf(v.x * scale) * post;
    v.y = rintf(v.y * scale) * post;
    v.z = rintf(v.z * scale) * post;
    v.w = rintf(v.w * scale) * post;
    *(float4*)(y + i) = v;
}

extern "C" void kernel_launch(void* const* d_in, const int* in_sizes, int n_in,
                              void* d_out, int out_size, void* d_ws, size_t ws_size,
                              hipStream_t stream) {
    const float* x    = (const float*)d_in[0];
    const float* ln_w = (const float*)d_in[1];
    const float* ln_b = (const float*)d_in[2];
    const float* W    = (const float*)d_in[3];
    const float* b    = (const float*)d_in[4];
    float* out = (float*)d_out;

    unsigned char* ws = (unsigned char*)d_ws;
    unsigned* absmax  = (unsigned*)ws;                       // 4 B
    float*    frob    = (float*)(ws + 64);                   // 4 B
    float*    partial = (float*)(ws + 256);                  // 16 KB
    _Float16* Bh      = (_Float16*)(ws + (size_t)(1u << 20));                       // 32 MiB
    _Float16* Af      = (_Float16*)(ws + (size_t)(1u << 20) + ((size_t)32u << 20)); // 128 MiB

    init_kernel<<<1, 1, 0, stream>>>(absmax);
    prep_kernel<<<4096, 256, 0, stream>>>(W, Bh, partial);
    frob_reduce_kernel<<<1, 256, 0, stream>>>(partial, frob);
    ln_kernel<<<NROWS, 256, 0, stream>>>(x, ln_w, ln_b, Af);
    gemm_kernel<<<dim3(DIM / BN, NROWS / BM), 256, 0, stream>>>(Af, Bh, b, out, absmax);
    finalize_kernel<<<(NROWS * (size_t)DIM) / (256 * 4), 256, 0, stream>>>(out, absmax, frob);
}
